// LSTMDecoder_5669356832362
// MI455X (gfx1250) — matrix operand order repaired and checked by reading, not compiled, not run on hardware
//
#include <hip/hip_runtime.h>
#include <hip/hip_bf16.h>
#include <math.h>

typedef __bf16 bf16;
typedef __attribute__((ext_vector_type(16))) __bf16 v16bf;
typedef __attribute__((ext_vector_type(8)))  float  v8f;

#define DEV __device__ __forceinline__

namespace {
constexpr int BB   = 128;    // sentences
constexpr int LL   = 128;    // seq len
constexpr int DIN  = 1024;
constexpr int DH   = 1024;
constexpr int G4   = 4096;   // 4*DH
constexpr int DOUT = 128;
constexpr int DEMB = 128;
constexpr int NTOK = BB * LL;        // 16384
constexpr int WIH_K = DIN + DEMB;    // 1152
}

union FragU { v16bf v; uint4 q[2]; };

// A-matrix 16x32 bf16 fragment (ISA 7.12.2): lane<16 holds row=lane,
// elems 0..7 = K 0..7, elems 8..15 = K 16..23; lane>=16 holds K 8..15 / 24..31.
DEV v16bf load_a_frag(const bf16* base, int ld, int lane) {
  const bf16* p = base + (size_t)(lane & 15) * ld + ((lane >> 4) << 3);
  FragU f;
  f.q[0] = *reinterpret_cast<const uint4*>(p);       // K = kb+0..7
  f.q[1] = *reinterpret_cast<const uint4*>(p + 16);  // K = kb+16..23
  return f.v;
}

// B-matrix 32x16 bf16 fragment: B[k][n] = W[n0+n][k0+k] for row-major W[N,K].
// lane<16: col=lane, elems 0..7 = K 0..7, elems 8..15 = K 16..23; lane>=16: K 8..15 / 24..31.
DEV v16bf load_b_frag(const bf16* base, int ld, int lane) {
  const bf16* p = base + (size_t)(lane & 15) * ld + ((lane >> 4) << 3);
  FragU f;
  f.q[0] = *reinterpret_cast<const uint4*>(p);       // K = kb+0..7
  f.q[1] = *reinterpret_cast<const uint4*>(p + 16);  // K = kb+16..23
  return f.v;
}

DEV v8f wmma_bf16(v16bf a, v16bf b, v8f c) {
  return __builtin_amdgcn_wmma_f32_16x16x32_bf16(false, a, false, b,
                                                 (short)0, c, false, false);
}

DEV float sigmoidf_(float x) { return 1.0f / (1.0f + __expf(-x)); }

// ---------------- prep kernels ----------------

__global__ void k_cvt(const float* __restrict__ s, bf16* __restrict__ d, int n) {
  int i = blockIdx.x * blockDim.x + threadIdx.x;
  if (i < n) d[i] = (bf16)s[i];
}

__global__ void k_split_wih(const float* __restrict__ wih,
                            bf16* __restrict__ w1, bf16* __restrict__ w2) {
  int i = blockIdx.x * blockDim.x + threadIdx.x;
  if (i >= G4 * WIH_K) return;
  int j = i / WIH_K, k = i - j * WIH_K;
  bf16 v = (bf16)wih[i];
  if (k < DIN) w1[(size_t)j * DIN + k] = v;
  else         w2[(size_t)j * DEMB + (k - DIN)] = v;
}

__global__ void k_init_state(const float* __restrict__ init,
                             bf16* __restrict__ prev, bf16* __restrict__ h0,
                             float* __restrict__ c0) {
  int i = blockIdx.x * blockDim.x + threadIdx.x;
  if (i < BB * DEMB) prev[i] = (bf16)init[i & (DEMB - 1)];
  if (i < BB * DH) { h0[i] = (bf16)0.0f; c0[i] = 0.0f; }
}

// ---------------- big parallel GEMM: Xpre = enc @ W1^T + (b_ih+b_hh) -------
// one wave = 32 rows x 64 cols (8 accumulators; 2 A frags x 4 B frags),
// software-pipelined: next k-step's fragments load during current WMMAs.

__global__ void k_pregemm(const bf16* __restrict__ X, const bf16* __restrict__ W1,
                          const float* __restrict__ bih, const float* __restrict__ bhh,
                          float* __restrict__ Xpre) {
  int lane = threadIdx.x & 31;
  int wid  = blockIdx.x * (blockDim.x >> 5) + (threadIdx.x >> 5);
  int mt = wid >> 6;             // 0..511  (32-row tile)
  int ng = wid & 63;             // 0..63   (64-col group)
  int m0 = mt * 32, n0 = ng * 64;
  const bf16* A0 = X + (size_t)m0 * DIN;
  const bf16* A1 = A0 + (size_t)16 * DIN;
  const bf16* Wb = W1 + (size_t)n0 * DIN;

  v8f acc[8] = {};
  v16bf a0 = load_a_frag(A0, DIN, lane);
  v16bf a1 = load_a_frag(A1, DIN, lane);
  v16bf b[4];
#pragma unroll
  for (int j = 0; j < 4; ++j)
    b[j] = load_b_frag(Wb + (size_t)j * 16 * DIN, DIN, lane);

  for (int kk = 0; kk < DIN - 32; kk += 32) {
    v16bf a0n = load_a_frag(A0 + kk + 32, DIN, lane);
    v16bf a1n = load_a_frag(A1 + kk + 32, DIN, lane);
    v16bf bn[4];
#pragma unroll
    for (int j = 0; j < 4; ++j)
      bn[j] = load_b_frag(Wb + (size_t)j * 16 * DIN + kk + 32, DIN, lane);
#pragma unroll
    for (int j = 0; j < 4; ++j) {
      acc[j]     = wmma_bf16(a0, b[j], acc[j]);
      acc[4 + j] = wmma_bf16(a1, b[j], acc[4 + j]);
    }
    a0 = a0n; a1 = a1n;
#pragma unroll
    for (int j = 0; j < 4; ++j) b[j] = bn[j];
  }
#pragma unroll
  for (int j = 0; j < 4; ++j) {             // peeled last k-step
    acc[j]     = wmma_bf16(a0, b[j], acc[j]);
    acc[4 + j] = wmma_bf16(a1, b[j], acc[4 + j]);
  }

#pragma unroll
  for (int r = 0; r < 2; ++r) {
#pragma unroll
    for (int j = 0; j < 4; ++j) {
      int col = n0 + j * 16 + (lane & 15);
      float bias = bih[col] + bhh[col];
#pragma unroll
      for (int e = 0; e < 8; ++e) {
        int row = m0 + r * 16 + ((lane >> 4) << 3) + e;       // C-layout (7.12.2)
        Xpre[(size_t)row * G4 + col] = acc[4 * r + j][e] + bias;
      }
    }
  }
}

// ---------------- per-timestep recurrent gate + cell kernel ----------------
// one wave owns a 16x16 (batch, hidden) tile; 4 accumulators = gates i,f,g,o.
// Pipelined across the W_hh -> W_emb boundary.

__global__ void k_lstm_step(const bf16* __restrict__ hin, const bf16* __restrict__ prev,
                            const bf16* __restrict__ Whh, const bf16* __restrict__ W2,
                            const float* __restrict__ Xpre, float* __restrict__ cbuf,
                            bf16* __restrict__ hout, int t) {
  int lane = threadIdx.x & 31;
  int wid  = blockIdx.x * (blockDim.x >> 5) + (threadIdx.x >> 5);
  int bt = wid >> 6;             // 0..7  batch tile
  int jt = wid & 63;             // 0..63 hidden tile
  int m0 = bt * 16, j0 = jt * 16;
  const bf16* Ah = hin + (size_t)m0 * DH;
  const bf16* Ae = prev + (size_t)m0 * DEMB;

  v8f acc[4] = {};
  v16bf a = load_a_frag(Ah, DH, lane);
  v16bf b[4];
#pragma unroll
  for (int g = 0; g < 4; ++g)
    b[g] = load_b_frag(Whh + (size_t)(g * DH + j0) * DH, DH, lane);

  for (int kk = 0; kk < DH - 32; kk += 32) {                  // h @ W_hh^T
    v16bf an = load_a_frag(Ah + kk + 32, DH, lane);
    v16bf bn[4];
#pragma unroll
    for (int g = 0; g < 4; ++g)
      bn[g] = load_b_frag(Whh + (size_t)(g * DH + j0) * DH + kk + 32, DH, lane);
#pragma unroll
    for (int g = 0; g < 4; ++g) acc[g] = wmma_bf16(a, b[g], acc[g]);
    a = an;
#pragma unroll
    for (int g = 0; g < 4; ++g) b[g] = bn[g];
  }
  {                                                           // bridge to emb part
    v16bf an = load_a_frag(Ae, DEMB, lane);
    v16bf bn[4];
#pragma unroll
    for (int g = 0; g < 4; ++g)
      bn[g] = load_b_frag(W2 + (size_t)(g * DH + j0) * DEMB, DEMB, lane);
#pragma unroll
    for (int g = 0; g < 4; ++g) acc[g] = wmma_bf16(a, b[g], acc[g]);
    a = an;
#pragma unroll
    for (int g = 0; g < 4; ++g) b[g] = bn[g];
  }
  for (int kk = 0; kk < DEMB - 32; kk += 32) {                // prev_emb @ W_emb^T
    v16bf an = load_a_frag(Ae + kk + 32, DEMB, lane);
    v16bf bn[4];
#pragma unroll
    for (int g = 0; g < 4; ++g)
      bn[g] = load_b_frag(W2 + (size_t)(g * DH + j0) * DEMB + kk + 32, DEMB, lane);
#pragma unroll
    for (int g = 0; g < 4; ++g) acc[g] = wmma_bf16(a, b[g], acc[g]);
    a = an;
#pragma unroll
    for (int g = 0; g < 4; ++g) b[g] = bn[g];
  }
#pragma unroll
  for (int g = 0; g < 4; ++g) acc[g] = wmma_bf16(a, b[g], acc[g]);  // peeled tail

  int col = j0 + (lane & 15);
#pragma unroll
  for (int e = 0; e < 8; ++e) {
    int brow = m0 + ((lane >> 4) << 3) + e;
    size_t xr = ((size_t)brow * LL + t) * (size_t)G4;         // Xpre row = b*L + t
    float gi = acc[0][e] + Xpre[xr + col];
    float gf = acc[1][e] + Xpre[xr + DH + col];
    float gg = acc[2][e] + Xpre[xr + 2 * DH + col];
    float go = acc[3][e] + Xpre[xr + 3 * DH + col];
    size_t ci = (size_t)brow * DH + col;
    float cn = sigmoidf_(gf) * cbuf[ci] + sigmoidf_(gi) * tanhf(gg);
    cbuf[ci] = cn;
    hout[ci] = (bf16)(sigmoidf_(go) * tanhf(cn));
  }
}

// ---------------- per-timestep logits + argmax + embedding feedback --------
// single block, 8 waves; wave w computes rows w*16..w*16+15 x all 128 cols

__global__ void k_logits_step(const bf16* __restrict__ h, const bf16* __restrict__ Wout,
                              const float* __restrict__ bout, const bf16* __restrict__ Emb,
                              bf16* __restrict__ prev, float* __restrict__ out, int t) {
  __shared__ int sidx[BB];
  int lane = threadIdx.x & 31;
  int w    = threadIdx.x >> 5;       // 0..7
  int m0   = w * 16;
  const bf16* Ah = h + (size_t)m0 * DH;

  v8f acc[8] = {};
  v16bf a = load_a_frag(Ah, DH, lane);
  v16bf b[8];
#pragma unroll
  for (int nt = 0; nt < 8; ++nt)
    b[nt] = load_b_frag(Wout + (size_t)(nt * 16) * DH, DH, lane);

  for (int kk = 0; kk < DH - 32; kk += 32) {
    v16bf an = load_a_frag(Ah + kk + 32, DH, lane);
    v16bf bn[8];
#pragma unroll
    for (int nt = 0; nt < 8; ++nt)
      bn[nt] = load_b_frag(Wout + (size_t)(nt * 16) * DH + kk + 32, DH, lane);
#pragma unroll
    for (int nt = 0; nt < 8; ++nt) acc[nt] = wmma_bf16(a, b[nt], acc[nt]);
    a = an;
#pragma unroll
    for (int nt = 0; nt < 8; ++nt) b[nt] = bn[nt];
  }
#pragma unroll
  for (int nt = 0; nt < 8; ++nt) acc[nt] = wmma_bf16(a, b[nt], acc[nt]);

  float bestv[8];
  int   besti[8];
#pragma unroll
  for (int e = 0; e < 8; ++e) { bestv[e] = -3.4e38f; besti[e] = 0x7fffffff; }
#pragma unroll
  for (int nt = 0; nt < 8; ++nt) {
    int col = nt * 16 + (lane & 15);
#pragma unroll
    for (int e = 0; e < 8; ++e) {
      int brow = m0 + ((lane >> 4) << 3) + e;
      float v = acc[nt][e] + bout[col];
      out[((size_t)brow * LL + t) * DOUT + col] = v;           // out[b, t, :]
      if (v > bestv[e] || (v == bestv[e] && col < besti[e])) { bestv[e] = v; besti[e] = col; }
    }
  }
#pragma unroll
  for (int e = 0; e < 8; ++e) {
#pragma unroll
    for (int off = 8; off >= 1; off >>= 1) {                   // reduce 16-lane halves
      float ov = __shfl_xor(bestv[e], off);
      int   oi = __shfl_xor(besti[e], off);
      if (ov > bestv[e] || (ov == bestv[e] && oi < besti[e])) { bestv[e] = ov; besti[e] = oi; }
    }
    if ((lane & 15) == 0) sidx[m0 + ((lane >> 4) << 3) + e] = besti[e];
  }
  __syncthreads();
  for (int i = threadIdx.x; i < BB * DEMB; i += blockDim.x)    // embedding gather
    prev[i] = Emb[(size_t)sidx[i >> 7] * DEMB + (i & 127)];
}

// ---------------- launch ----------------

extern "C" void kernel_launch(void* const* d_in, const int* in_sizes, int n_in,
                              void* d_out, int out_size, void* d_ws, size_t ws_size,
                              hipStream_t stream) {
  (void)in_sizes; (void)n_in; (void)out_size; (void)ws_size;
  const float* enc  = (const float*)d_in[0];
  const float* wih  = (const float*)d_in[1];
  const float* whh  = (const float*)d_in[2];
  const float* bih  = (const float*)d_in[3];
  const float* bhh  = (const float*)d_in[4];
  const float* wout = (const float*)d_in[5];
  const float* bout = (const float*)d_in[6];
  const float* emb  = (const float*)d_in[7];
  const float* init = (const float*)d_in[8];
  float* out = (float*)d_out;

  char* p = (char*)d_ws;
  auto take = [&](size_t bytes) { char* r = p; p += (bytes + 255) & ~(size_t)255; return r; };
  float* Xpre = (float*)take((size_t)NTOK * G4 * 4);   // precomputed x-part + bias
  bf16*  Xbf  = (bf16*) take((size_t)NTOK * DIN * 2);
  bf16*  W1   = (bf16*) take((size_t)G4 * DIN * 2);    // W_ih[:, :1024]
  bf16*  W2   = (bf16*) take((size_t)G4 * DEMB * 2);   // W_ih[:, 1024:]
  bf16*  Whh  = (bf16*) take((size_t)G4 * DH * 2);
  bf16*  Wo   = (bf16*) take((size_t)DOUT * DH * 2);
  bf16*  Embb = (bf16*) take((size_t)DOUT * DEMB * 2);
  bf16*  hA   = (bf16*) take((size_t)BB * DH * 2);
  bf16*  hB   = (bf16*) take((size_t)BB * DH * 2);
  float* cbuf = (float*)take((size_t)BB * DH * 4);
  bf16*  prev = (bf16*) take((size_t)BB * DEMB * 2);

  // precision conversions (fp32 -> bf16) + state init
  k_cvt<<<(NTOK * DIN + 255) / 256, 256, 0, stream>>>(enc, Xbf, NTOK * DIN);
  k_cvt<<<(G4 * DH + 255) / 256, 256, 0, stream>>>(whh, Whh, G4 * DH);
  k_cvt<<<(DOUT * DH + 255) / 256, 256, 0, stream>>>(wout, Wo, DOUT * DH);
  k_cvt<<<(DOUT * DEMB + 255) / 256, 256, 0, stream>>>(emb, Embb, DOUT * DEMB);
  k_split_wih<<<(G4 * WIH_K + 255) / 256, 256, 0, stream>>>(wih, W1, W2);
  k_init_state<<<(BB * DH + 255) / 256, 256, 0, stream>>>(init, prev, hA, cbuf);

  // hoisted, fully parallel input GEMM: [16384x1024] x [1024x4096], 32x64/wave
  k_pregemm<<<(NTOK / 32) * (G4 / 64) / 8, 256, 0, stream>>>(Xbf, W1, bih, bhh, Xpre);

  // sequential recurrence: 2 kernels per timestep, h double-buffered
  for (int t = 0; t < LL; ++t) {
    const bf16* hin  = (t & 1) ? hB : hA;
    bf16*       hout = (t & 1) ? hA : hB;
    k_lstm_step<<<64, 256, 0, stream>>>(hin, prev, Whh, W2, Xpre, cbuf, hout, t);
    k_logits_step<<<1, 256, 0, stream>>>(hout, Wo, bout, Embb, prev, out, t);
  }
}
